// MultiHead_Attention_Lattice_rel_55121610277409
// MI455X (gfx1250) — matrix operand order repaired and checked by reading, not compiled, not run on hardware
//
#include <hip/hip_runtime.h>

// ---------------------------------------------------------------------------
// Problem constants (from reference)
// ---------------------------------------------------------------------------
#define Bv      2
#define Lv      192
#define Hv      512
#define NHv     8
#define DPHv    64
#define MAXLENv 512
#define PEROWS  1025      // 2*MAXLEN+1
#define BL      (Bv*Lv)   // 384

typedef _Float16 f16;
typedef _Float16 v16h __attribute__((ext_vector_type(16)));
typedef _Float16 v8h  __attribute__((ext_vector_type(8)));
typedef float    v8f  __attribute__((ext_vector_type(8)));

// Build a 16-half WMMA operand from two contiguous 8-half (16B) chunks.
__device__ __forceinline__ v16h ld16(const f16* __restrict__ p0,
                                     const f16* __restrict__ p1) {
  v8h lo = *(const v8h*)p0;
  v8h hi = *(const v8h*)p1;
  v16h r;
#pragma unroll
  for (int i = 0; i < 8; ++i) { r[i] = lo[i]; r[i + 8] = hi[i]; }
  return r;
}

__device__ __forceinline__ v8f wmma_f16(v16h a, v16h b, v8f c) {
  return __builtin_amdgcn_wmma_f32_16x16x32_f16(false, a, false, b,
                                                (short)0, c, false, false);
}

// ---------------------------------------------------------------------------
// Elementwise helpers
// ---------------------------------------------------------------------------
__global__ void k_cvt_f32_f16(const float* __restrict__ in, f16* __restrict__ out, int n) {
  int i = blockIdx.x * blockDim.x + threadIdx.x;
  if (i < n) out[i] = (f16)in[i];
}

// WrT[f,h] = Wr[h,f]  (f16)
__global__ void k_transpose_wr(const float* __restrict__ Wr, f16* __restrict__ WrT) {
  int i = blockIdx.x * blockDim.x + threadIdx.x;
  if (i >= Hv * Hv) return;
  int h = i / Hv, f = i % Hv;
  WrT[(long)f * Hv + h] = (f16)Wr[(long)h * Hv + f];
}

// qu = q + u_bias(flat), g = q + v_bias(flat)   [BL, H] each, f16
__global__ void k_qu_g(const float* __restrict__ qproj,
                       const float* __restrict__ uflat,
                       const float* __restrict__ vflat,
                       f16* __restrict__ qu16, f16* __restrict__ g16) {
  int i = blockIdx.x * blockDim.x + threadIdx.x;
  if (i >= BL * Hv) return;
  int h = i % Hv;
  float q = qproj[i];
  qu16[i] = (f16)(q + uflat[h]);
  g16[i]  = (f16)(q + vflat[h]);
}

// sconst[b,q,n] = sum_d (qproj + v_bias)[n*64+d] * br[n*64+d]
__global__ void k_sconst(const float* __restrict__ qproj,
                         const float* __restrict__ vflat,
                         const float* __restrict__ br,
                         float* __restrict__ sc) {
  int i = blockIdx.x * blockDim.x + threadIdx.x;
  if (i >= BL * NHv) return;
  int n = i % NHv, bq = i / NHv;
  const float* qp = qproj + (long)bq * Hv + n * DPHv;
  const float* vb = vflat + n * DPHv;
  const float* bb = br + n * DPHv;
  float s = 0.f;
  for (int d = 0; d < DPHv; ++d) s += (qp[d] + vb[d]) * bb[d];
  sc[i] = s;
}

// vT[b,n,d,k] = vproj[b,k,n*64+d]  (f16)
__global__ void k_vtrans(const float* __restrict__ vproj, f16* __restrict__ vT) {
  int i = blockIdx.x * blockDim.x + threadIdx.x;
  if (i >= Bv * NHv * DPHv * Lv) return;
  int k = i % Lv;
  int d = (i / Lv) % DPHv;
  int n = (i / (Lv * DPHv)) % NHv;
  int b = i / (Lv * DPHv * NHv);
  vT[i] = (f16)vproj[((long)(b * Lv + k) * Hv) + n * DPHv + d];
}

// ---------------------------------------------------------------------------
// Generic WMMA GEMM:  C[M,N] = A[M,K] * BT[N,K]^T  (+ bias[N])
// A, BT are f16 row-major with K contiguous. One wave per 16x16 C tile.
// K must be a multiple of 32. M,N tails handled by clamp + guarded store.
// ---------------------------------------------------------------------------
__global__ void k_gemm_wmma(const f16* __restrict__ A, int lda,
                            const f16* __restrict__ BT, int ldb,
                            int M, int N, int K,
                            const float* __restrict__ bias,
                            float* __restrict__ Cf, int ldcf,
                            f16* __restrict__ Ch, int ldch) {
  const int lane = threadIdx.x & 31;
  const int wid  = (blockIdx.x * blockDim.x + threadIdx.x) >> 5;
  const int ntiles = (N + 15) >> 4;
  const int mtiles = (M + 15) >> 4;
  if (wid >= ntiles * mtiles) return;
  const int mt = wid / ntiles, nt = wid % ntiles;
  const int r = lane & 15;
  const int fsel = lane >> 4;
  int mrow = mt * 16 + r; if (mrow > M - 1) mrow = M - 1;   // clamp (safe addr)
  int nrow = nt * 16 + r; if (nrow > N - 1) nrow = N - 1;
  const f16* ap = A + (long)mrow * lda + fsel * 8;
  const f16* bp = BT + (long)nrow * ldb + fsel * 8;
  v8f acc = {};
  for (int kb = 0; kb < K; kb += 32) {
    v16h a = ld16(ap + kb, ap + kb + 16);
    v16h b = ld16(bp + kb, bp + kb + 16);
    acc = wmma_f16(a, b, acc);
  }
  const int ncol = nt * 16 + (lane & 15);
  if (ncol < N) {
    float bvv = bias ? bias[ncol] : 0.0f;
#pragma unroll
    for (int v = 0; v < 8; ++v) {
      int mr = mt * 16 + v + 8 * fsel;
      if (mr < M) {
        float val = acc[v] + bvv;
        if (Cf) Cf[(long)mr * ldcf + ncol] = val;
        if (Ch) Ch[(long)mr * ldch + ncol] = (f16)val;
      }
    }
  }
}

// ---------------------------------------------------------------------------
// Scores part 1:  sAC[b,n,q,k] = sum_d (q+u)[b,q,n,d] * k[b,k,n,d]
// One wave per (b, n, 16q-tile, 16k-tile).
// ---------------------------------------------------------------------------
__global__ void k_scores_ac(const f16* __restrict__ qu,   // [BL,H]
                            const f16* __restrict__ kp,   // [BL,H]
                            float* __restrict__ sAC) {    // [B,NH,L,L]
  const int lane = threadIdx.x & 31;
  const int wid  = (blockIdx.x * blockDim.x + threadIdx.x) >> 5;
  if (wid >= Bv * 12 * 12 * NHv) return;
  const int n  = wid & 7;
  const int t  = wid >> 3;          // 0..287
  const int kt = t % 12;
  const int qt = (t / 12) % 12;
  const int b  = t / 144;
  const int r = lane & 15, fsel = lane >> 4;
  const f16* ap = qu + ((long)(b * Lv + qt * 16 + r) * Hv) + n * DPHv + fsel * 8;
  const f16* bp = kp + ((long)(b * Lv + kt * 16 + r) * Hv) + n * DPHv + fsel * 8;
  v8f acc = {};
#pragma unroll
  for (int kb = 0; kb < DPHv; kb += 32) {
    v16h a = ld16(ap + kb, ap + kb + 16);
    v16h bm = ld16(bp + kb, bp + kb + 16);
    acc = wmma_f16(a, bm, acc);
  }
  const int kcol = kt * 16 + (lane & 15);
#pragma unroll
  for (int v = 0; v < 8; ++v) {
    int q = qt * 16 + v + 8 * fsel;
    sAC[(((long)(b * NHv + n) * Lv + q) * Lv) + kcol] = acc[v];
  }
}

// ---------------------------------------------------------------------------
// Scores part 2:  sBD[b,n,q,k] = sum_f relu1[b,q,k,f] * G[b,q,n,f]
//   relu1[f] = relu(T0[dss,f]+T1[dse,f]+T2[des,f]+T3[dee,f]+b_fus[f])
// One wave per (b,q, 16k-tile). A-operand (M=k rows, K=f) is built directly
// in the WMMA A register layout from table gathers; B = G[b,q] (N=heads,pad16).
// ---------------------------------------------------------------------------
__global__ void k_scores_bd(const float* __restrict__ T,     // [4,1025,512]
                            const float* __restrict__ bfus,  // [512]
                            const f16*  __restrict__ G,      // [BL,8,512]
                            const int*  __restrict__ pos_s,
                            const int*  __restrict__ pos_e,
                            float* __restrict__ sBD) {       // [B,NH,L,L]
  const int lane = threadIdx.x & 31;
  const int wid  = (blockIdx.x * blockDim.x + threadIdx.x) >> 5;
  if (wid >= BL * 12) return;
  const int kt = wid % 12;
  const int bq = wid / 12;
  const int b = bq / Lv, q = bq % Lv;
  const int m = lane & 15, fsel = lane >> 4;
  const int k = kt * 16 + m;
  const int psq = pos_s[b * Lv + q], peq = pos_e[b * Lv + q];
  const int psk = pos_s[b * Lv + k], pek = pos_e[b * Lv + k];
  int dss = psq - psk + MAXLENv;
  int dse = psq - pek + MAXLENv;
  int des = peq - psk + MAXLENv;
  int dee = peq - pek + MAXLENv;
  dss = dss < 0 ? 0 : (dss > 1024 ? 1024 : dss);
  dse = dse < 0 ? 0 : (dse > 1024 ? 1024 : dse);
  des = des < 0 ? 0 : (des > 1024 ? 1024 : des);
  dee = dee < 0 ? 0 : (dee > 1024 ? 1024 : dee);
  const float* t0 = T + (long)dss * Hv;
  const float* t1 = T + (long)(PEROWS + dse) * Hv;
  const float* t2 = T + (long)(2 * PEROWS + des) * Hv;
  const float* t3 = T + (long)(3 * PEROWS + dee) * Hv;
  int nb = lane & 15; if (nb > 7) nb = 7;                    // pad heads 8..15
  const f16* bp = G + ((long)bq * NHv + nb) * Hv + fsel * 8;
  v8f acc = {};
  for (int kb = 0; kb < Hv; kb += 32) {
    v16h a;
#pragma unroll
    for (int c = 0; c < 2; ++c) {
      const int f = kb + fsel * 8 + c * 16;
#pragma unroll
      for (int i = 0; i < 8; ++i) {
        float s = t0[f + i] + t1[f + i] + t2[f + i] + t3[f + i] + bfus[f + i];
        a[c * 8 + i] = (f16)fmaxf(s, 0.0f);
      }
    }
    v16h bm = ld16(bp + kb, bp + kb + 16);
    acc = wmma_f16(a, bm, acc);
  }
  const int n = lane & 15;
  if (n < 8) {
#pragma unroll
    for (int v = 0; v < 8; ++v) {
      int kr = kt * 16 + v + 8 * fsel;
      sBD[(((long)(b * NHv + n) * Lv + q) * Lv) + kr] = acc[v];
    }
  }
}

// ---------------------------------------------------------------------------
// Softmax over k: one wave per (b,n,q); 192 = 6 x 32 lanes.
// ---------------------------------------------------------------------------
__global__ void k_softmax(const float* __restrict__ sAC,
                          const float* __restrict__ sBD,
                          const float* __restrict__ sconst,  // [BL, NH]
                          const int*   __restrict__ seq_len,
                          f16* __restrict__ attn) {          // [B,NH,L,L] f16
  const int lane = threadIdx.x & 31;
  const int wid  = (blockIdx.x * blockDim.x + threadIdx.x) >> 5;
  if (wid >= Bv * NHv * Lv) return;
  const int q = wid % Lv;
  const int n = (wid / Lv) % NHv;
  const int b = wid / (Lv * NHv);
  const float c = sconst[((long)(b * Lv + q)) * NHv + n];
  const int sl = seq_len[b];
  const long base = ((long)(b * NHv + n) * Lv + q) * Lv;
  float s[6];
  float mx = -3.0e38f;
#pragma unroll
  for (int j = 0; j < 6; ++j) {
    int k = lane + j * 32;
    float v = (sAC[base + k] + sBD[base + k] + c) * 0.125f;  // 1/sqrt(64)
    v = (k < sl) ? v : -1.0e15f;
    s[j] = v;
    mx = fmaxf(mx, v);
  }
  for (int off = 16; off > 0; off >>= 1) mx = fmaxf(mx, __shfl_xor(mx, off, 32));
  float sum = 0.f;
#pragma unroll
  for (int j = 0; j < 6; ++j) { s[j] = __expf(s[j] - mx); sum += s[j]; }
  for (int off = 16; off > 0; off >>= 1) sum += __shfl_xor(sum, off, 32);
  float inv = 1.0f / sum;
#pragma unroll
  for (int j = 0; j < 6; ++j) attn[base + lane + j * 32] = (f16)(s[j] * inv);
}

// ---------------------------------------------------------------------------
// AV: obuf[b,q,n*64+d] = sum_k attn[b,n,q,k] * vT[b,n,d,k]
// One wave per (b, n, 16q-tile, 16d-tile).
// ---------------------------------------------------------------------------
__global__ void k_av(const f16* __restrict__ attn,   // [B,NH,L,L]
                     const f16* __restrict__ vT,     // [B,NH,64,L]
                     f16* __restrict__ obuf) {       // [BL,H]
  const int lane = threadIdx.x & 31;
  const int wid  = (blockIdx.x * blockDim.x + threadIdx.x) >> 5;
  if (wid >= Bv * NHv * 12 * 4) return;
  const int dt = wid & 3;
  const int qt = (wid >> 2) % 12;
  const int n  = (wid / 48) % NHv;
  const int b  = wid / (48 * NHv);
  const int r = lane & 15, fsel = lane >> 4;
  const f16* ap = attn + (((long)(b * NHv + n) * Lv + qt * 16 + r) * Lv) + fsel * 8;
  const f16* bp = vT + (((long)(b * NHv + n) * DPHv + dt * 16 + r) * Lv) + fsel * 8;
  v8f acc = {};
#pragma unroll
  for (int kb = 0; kb < Lv; kb += 32) {
    v16h a = ld16(ap + kb, ap + kb + 16);
    v16h bm = ld16(bp + kb, bp + kb + 16);
    acc = wmma_f16(a, bm, acc);
  }
  const int dcol = dt * 16 + (lane & 15);
#pragma unroll
  for (int v = 0; v < 8; ++v) {
    int q = qt * 16 + v + 8 * fsel;
    obuf[((long)(b * Lv + q) * Hv) + n * DPHv + dcol] = (f16)acc[v];
  }
}

// ---------------------------------------------------------------------------
// Host launcher
// ---------------------------------------------------------------------------
extern "C" void kernel_launch(void* const* d_in, const int* in_sizes, int n_in,
                              void* d_out, int out_size, void* d_ws, size_t ws_size,
                              hipStream_t stream) {
  (void)in_sizes; (void)n_in; (void)out_size; (void)ws_size;
  const float* key   = (const float*)d_in[0];
  const float* query = (const float*)d_in[1];
  const float* value = (const float*)d_in[2];
  const int*   seqln = (const int*)d_in[3];
  const int*   pos_s = (const int*)d_in[5];
  const int*   pos_e = (const int*)d_in[6];
  const float* pe    = (const float*)d_in[7];
  const float* W_fus = (const float*)d_in[8];
  const float* b_fus = (const float*)d_in[9];
  const float* Wk    = (const float*)d_in[10];
  const float* bk    = (const float*)d_in[11];
  const float* Wq    = (const float*)d_in[12];
  const float* bq    = (const float*)d_in[13];
  const float* Wv    = (const float*)d_in[14];
  const float* bv    = (const float*)d_in[15];
  const float* Wr    = (const float*)d_in[16];
  const float* br    = (const float*)d_in[17];
  const float* u_b   = (const float*)d_in[18];   // [8,64] -> flat 512
  const float* v_b   = (const float*)d_in[19];
  const float* Wff   = (const float*)d_in[20];
  const float* bff   = (const float*)d_in[21];

  // ---- workspace carve (all chunks 256B-aligned sizes) ----
  char* w = (char*)d_ws;
  auto take = [&](size_t bytes) { char* p = w; w += (bytes + 255) & ~size_t(255); return p; };
  f16*   key16   = (f16*)take(BL * Hv * 2);
  f16*   qry16   = (f16*)take(BL * Hv * 2);
  f16*   val16   = (f16*)take(BL * Hv * 2);
  f16*   pe16    = (f16*)take((size_t)PEROWS * Hv * 2);
  f16*   wfus16  = (f16*)take((size_t)Hv * 1024 * 2);
  f16*   wk16    = (f16*)take((size_t)Hv * Hv * 2);
  f16*   wq16    = (f16*)take((size_t)Hv * Hv * 2);
  f16*   wv16    = (f16*)take((size_t)Hv * Hv * 2);
  f16*   wff16   = (f16*)take((size_t)Hv * Hv * 2);
  f16*   wrT16   = (f16*)take((size_t)Hv * Hv * 2);
  float* qproj   = (float*)take(BL * Hv * 4);
  float* vproj   = (float*)take(BL * Hv * 4);
  f16*   kproj16 = (f16*)take(BL * Hv * 2);
  f16*   qu16    = (f16*)take(BL * Hv * 2);
  f16*   g16     = (f16*)take(BL * Hv * 2);
  float* Ttab    = (float*)take((size_t)4 * PEROWS * Hv * 4);
  f16*   G16     = (f16*)take((size_t)BL * NHv * Hv * 2);
  float* sconst  = (float*)take((size_t)BL * NHv * 4);
  float* sAC     = (float*)take((size_t)Bv * NHv * Lv * Lv * 4);
  float* sBD     = (float*)take((size_t)Bv * NHv * Lv * Lv * 4);
  f16*   attn16  = (f16*)take((size_t)Bv * NHv * Lv * Lv * 2);
  f16*   vT16    = (f16*)take((size_t)Bv * NHv * DPHv * Lv * 2);
  f16*   obuf16  = (f16*)take((size_t)BL * Hv * 2);

  const int TPB = 256;
  auto eb = [](int n) { return (n + 255) / 256; };
  auto gb = [](int tiles) { return (tiles + 7) / 8; };   // 8 waves/block

  // ---- 1. f32 -> f16 conversions ----
  k_cvt_f32_f16<<<eb(BL * Hv), TPB, 0, stream>>>(key,   key16, BL * Hv);
  k_cvt_f32_f16<<<eb(BL * Hv), TPB, 0, stream>>>(query, qry16, BL * Hv);
  k_cvt_f32_f16<<<eb(BL * Hv), TPB, 0, stream>>>(value, val16, BL * Hv);
  k_cvt_f32_f16<<<eb(PEROWS * Hv), TPB, 0, stream>>>(pe, pe16, PEROWS * Hv);
  k_cvt_f32_f16<<<eb(Hv * 1024), TPB, 0, stream>>>(W_fus, wfus16, Hv * 1024);
  k_cvt_f32_f16<<<eb(Hv * Hv), TPB, 0, stream>>>(Wk, wk16, Hv * Hv);
  k_cvt_f32_f16<<<eb(Hv * Hv), TPB, 0, stream>>>(Wq, wq16, Hv * Hv);
  k_cvt_f32_f16<<<eb(Hv * Hv), TPB, 0, stream>>>(Wv, wv16, Hv * Hv);
  k_cvt_f32_f16<<<eb(Hv * Hv), TPB, 0, stream>>>(Wff, wff16, Hv * Hv);
  k_transpose_wr<<<eb(Hv * Hv), TPB, 0, stream>>>(Wr, wrT16);

  // ---- 2. projections (WMMA): M=384, N=512, K=512 -> 768 tiles ----
  k_gemm_wmma<<<gb(24 * 32), TPB, 0, stream>>>(qry16, Hv, wq16, Hv, BL, Hv, Hv,
                                               bq, qproj, Hv, nullptr, 0);
  k_gemm_wmma<<<gb(24 * 32), TPB, 0, stream>>>(key16, Hv, wk16, Hv, BL, Hv, Hv,
                                               bk, nullptr, 0, kproj16, Hv);
  k_gemm_wmma<<<gb(24 * 32), TPB, 0, stream>>>(val16, Hv, wv16, Hv, BL, Hv, Hv,
                                               bv, vproj, Hv, nullptr, 0);

  // ---- 3. qu/g, sconst, vT ----
  k_qu_g<<<eb(BL * Hv), TPB, 0, stream>>>(qproj, u_b, v_b, qu16, g16);
  k_sconst<<<eb(BL * NHv), TPB, 0, stream>>>(qproj, v_b, br, sconst);
  k_vtrans<<<eb(Bv * NHv * DPHv * Lv), TPB, 0, stream>>>(vproj, vT16);

  // ---- 4. T tables: 4 x GEMM [1025 x 512 x K=256] -> 65*32 tiles each ----
  for (int j = 0; j < 4; ++j) {
    k_gemm_wmma<<<gb(65 * 32), TPB, 0, stream>>>(
        pe16, Hv, wfus16 + j * 256, 1024, PEROWS, Hv, 256,
        nullptr, Ttab + (size_t)j * PEROWS * Hv, Hv, nullptr, 0);
  }

  // ---- 5. G per head: M=384, N=512, K=64 ----
  for (int n = 0; n < NHv; ++n) {
    k_gemm_wmma<<<gb(24 * 32), TPB, 0, stream>>>(
        g16 + n * DPHv, Hv, wrT16 + n * DPHv, Hv, BL, Hv, DPHv,
        nullptr, nullptr, 0, G16 + n * Hv, NHv * Hv);
  }

  // ---- 6. scores ----
  k_scores_ac<<<(Bv * 12 * 12 * NHv + 7) / 8, TPB, 0, stream>>>(qu16, kproj16, sAC);
  k_scores_bd<<<(BL * 12 + 7) / 8, TPB, 0, stream>>>(Ttab, b_fus, G16, pos_s, pos_e, sBD);

  // ---- 7. softmax ----
  k_softmax<<<(Bv * NHv * Lv + 7) / 8, TPB, 0, stream>>>(sAC, sBD, sconst, seqln, attn16);

  // ---- 8. attn @ V ----
  k_av<<<(Bv * NHv * 12 * 4 + 7) / 8, TPB, 0, stream>>>(attn16, vT16, obuf16);

  // ---- 9. final projection to d_out (f32) ----
  k_gemm_wmma<<<gb(24 * 32), TPB, 0, stream>>>(obuf16, Hv, wff16, Hv, BL, Hv, Hv,
                                               bff, (float*)d_out, Hv, nullptr, 0);
}
